// CIE_Net_18829136625933
// MI455X (gfx1250) — matrix operand order repaired and checked by reading, not compiled, not run on hardware
//
#include <hip/hip_runtime.h>
#include <utility>

typedef __attribute__((ext_vector_type(16))) __bf16 v16bf;
typedef __attribute__((ext_vector_type(8)))  __bf16 v8bf;
typedef __attribute__((ext_vector_type(8)))  float  v8f;

#define NEGV (-1e30f)
#define SK_INV_TAU (20.0f)

// ---------------------------------------------------------------------------
// gfx1250 async global->LDS copy (ASYNCcnt-tracked, no VGPR round-trip).
// Inline asm per bridge doc (portable across toolchain builtin arities).
// LDS dest: generic pointers to __shared__ carry the LDS offset in addr[31:0].
// ---------------------------------------------------------------------------
__device__ __forceinline__ void async_copy_b128(const void* gsrc, void* ldst) {
  const unsigned int loff = (unsigned int)(unsigned long long)ldst;
  asm volatile("global_load_async_to_lds_b128 %0, %1, off"
               :: "v"(loff), "v"(gsrc)
               : "memory");
}
__device__ __forceinline__ void s_wait_async0() {
#if __has_builtin(__builtin_amdgcn_s_wait_asynccnt)
  __builtin_amdgcn_s_wait_asynccnt(0);
#else
  asm volatile("s_wait_asynccnt 0x0" ::: "memory");
#endif
}

// ---------------------------------------------------------------------------
// WMMA fragment loaders (ISA 7.12.2, wave32, 16-bit A 16x32 / B 32x16)
// A: lane half h, row m=lane&15: K = {8h..8h+7} U {16+8h..16+8h+7}
// B: lane half h, col n=lane&15: K = {8h..8h+7} U {16+8h..16+8h+7}
// B-operands are stored pre-transposed as [N][K] so both loaders read
// contiguous 16B chunks from LDS (ds_load_b128).
// ---------------------------------------------------------------------------
__device__ __forceinline__ v16bf load_fragA(const __bf16* row, int half) {
  v8bf lo = *(const v8bf*)(row + half * 8);
  v8bf hi = *(const v8bf*)(row + 16 + half * 8);
  return __builtin_shufflevector(lo, hi, 0,1,2,3,4,5,6,7,8,9,10,11,12,13,14,15);
}
__device__ __forceinline__ v16bf load_fragB(const __bf16* row, int half) {
  v8bf lo = *(const v8bf*)(row + half * 8);
  v8bf hi = *(const v8bf*)(row + 16 + half * 8);
  return __builtin_shufflevector(lo, hi, 0,1,2,3,4,5,6,7,8,9,10,11,12,13,14,15);
}

// ---------------------------------------------------------------------------
// Generic bf16 GEMM: C[M,N] = A[M,K] @ Bt[N,K]^T (+bias)(relu), f32 accumulate.
// Double-buffered LDS, async copies overlapped with WMMA compute.
// Optional batched operation via grid.z + element strides.
// ---------------------------------------------------------------------------
template<int BM, int BN, int WM, int WN>
__global__ void __launch_bounds__((BM / WM) * (BN / WN) * 32)
gemm_bf16_kernel(const __bf16* __restrict__ A, long long strideA,
                 const __bf16* __restrict__ Bt, long long strideB,
                 const float* __restrict__ bias,
                 float* __restrict__ Cf, __bf16* __restrict__ Cb, long long strideC,
                 int M, int N, int K, int relu)
{
  constexpr int KT = 32, KTP = 40;          // 80B row pitch: 16B aligned, bank-spread
  constexpr int WVM = BM / WM, WVN = BN / WN;
  constexpr int NT  = WVM * WVN * 32;
  constexpr int MT  = WM / 16, NTT = WN / 16;
  constexpr int CA  = (BM * 4) / NT, CB = (BN * 4) / NT;
  static_assert(CA * NT == BM * 4 && CB * NT == BN * 4, "exact staging trips");
  __shared__ __attribute__((aligned(16))) __bf16 As[2][BM][KTP];
  __shared__ __attribute__((aligned(16))) __bf16 Bs[2][BN][KTP];
  (void)M;
  const int z = blockIdx.z;
  A  += (size_t)z * strideA;
  Bt += (size_t)z * strideB;
  const int row0 = blockIdx.y * BM, col0 = blockIdx.x * BN;
  const int tid = threadIdx.x, lane = tid & 31, wave = tid >> 5;
  const int wm = (wave % WVM) * WM, wn = (wave / WVM) * WN;
  const int half = lane >> 4, rl = lane & 15;

  auto stage = [&](int k0, int p) {
#pragma unroll
    for (int t = 0; t < CA; ++t) {
      const int i = tid + t * NT, r = i >> 2, c = i & 3;
      async_copy_b128(A + (size_t)(row0 + r) * K + k0 + c * 8, &As[p][r][c * 8]);
    }
#pragma unroll
    for (int t = 0; t < CB; ++t) {
      const int i = tid + t * NT, r = i >> 2, c = i & 3;
      async_copy_b128(Bt + (size_t)(col0 + r) * K + k0 + c * 8, &Bs[p][r][c * 8]);
    }
  };

  const int nk = K / KT;
  stage(0, 0);
  v8f acc[MT][NTT] = {};
  for (int ki = 0; ki < nk; ++ki) {
    const int p = ki & 1;
    s_wait_async0();           // this wave's copies for buffer p have landed
    __syncthreads();           // ... and everyone else's
    if (ki + 1 < nk) stage((ki + 1) * KT, p ^ 1);   // overlap copy with compute
    v16bf af[MT], bfr[NTT];
#pragma unroll
    for (int tm = 0; tm < MT; ++tm) af[tm] = load_fragA(&As[p][wm + tm * 16 + rl][0], half);
#pragma unroll
    for (int tn = 0; tn < NTT; ++tn) bfr[tn] = load_fragB(&Bs[p][wn + tn * 16 + rl][0], half);
#pragma unroll
    for (int tm = 0; tm < MT; ++tm)
#pragma unroll
      for (int tn = 0; tn < NTT; ++tn)
        acc[tm][tn] = __builtin_amdgcn_wmma_f32_16x16x32_bf16(
            false, af[tm], false, bfr[tn], (short)0, acc[tm][tn], false, false);
    __syncthreads();           // all frag reads of buffer p done before re-stage
  }
  // C/D layout: VGPR r -> row (half*8 + r), col = lane&15
  const int rr = half * 8;
#pragma unroll
  for (int tm = 0; tm < MT; ++tm)
#pragma unroll
    for (int tn = 0; tn < NTT; ++tn) {
      const int gc = col0 + wn + tn * 16 + rl;
      const float bv = bias ? bias[gc] : 0.0f;
#pragma unroll
      for (int r = 0; r < 8; ++r) {
        const int gr = row0 + wm + tm * 16 + rr + r;
        float v = acc[tm][tn][r] + bv;
        if (relu) v = fmaxf(v, 0.0f);
        const size_t idx = (size_t)z * strideC + (size_t)gr * N + gc;
        if (Cf) Cf[idx] = v;
        if (Cb) Cb[idx] = (__bf16)v;
      }
    }
}

// ---------------------------------------------------------------------------
// Fused edge GEMM (layers 1,2): each block's 64 rows are the full j-range of
// one (b,i), so the per-channel message-passing reduction over j is done in
// LDS right after the GEMM. Emits relu(edge_x) bf16 (optional) and the new
// node embedding (relu(agg)+relu(sx)) as bf16. edge_x f32 never leaves LDS.
// ---------------------------------------------------------------------------
__global__ void __launch_bounds__(256)
edge_gemm_kernel(const __bf16* __restrict__ Ein, const __bf16* __restrict__ EwT,
                 const float* __restrict__ ebias, const float* __restrict__ Adj,
                 const float* __restrict__ nodex, const float* __restrict__ sx,
                 __bf16* __restrict__ e_out, __bf16* __restrict__ emb_out, int K)
{
  constexpr int BM = 64, BN = 128, WM = 32, WN = 32;
  constexpr int KT = 32, KTP = 40;
  constexpr int WVM = BM / WM;            // 2
  constexpr int MT = WM / 16, NTT = WN / 16;
  __shared__ __attribute__((aligned(16))) __bf16 As[2][BM][KTP];
  __shared__ __attribute__((aligned(16))) __bf16 Bs[2][BN][KTP];
  __shared__ float eds[64][BN + 1];

  const int row0 = blockIdx.y * BM, col0 = blockIdx.x * BN;
  const int tid = threadIdx.x, lane = tid & 31, wave = tid >> 5;
  const int wm = (wave % WVM) * WM, wn = (wave / WVM) * WN;
  const int half = lane >> 4, rl = lane & 15;

  auto stage = [&](int k0, int p) {
    {
      const int r = tid >> 2, c = tid & 3;     // 256 threads = 64 rows x 4 chunks
      async_copy_b128(Ein + (size_t)(row0 + r) * K + k0 + c * 8, &As[p][r][c * 8]);
    }
#pragma unroll
    for (int t = 0; t < 2; ++t) {
      const int i = tid + t * 256, r = i >> 2, c = i & 3;
      async_copy_b128(EwT + (size_t)(col0 + r) * K + k0 + c * 8, &Bs[p][r][c * 8]);
    }
  };

  const int nk = K / KT;
  stage(0, 0);
  v8f acc[MT][NTT] = {};
  for (int ki = 0; ki < nk; ++ki) {
    const int p = ki & 1;
    s_wait_async0();
    __syncthreads();
    if (ki + 1 < nk) stage((ki + 1) * KT, p ^ 1);
    v16bf af[MT], bfr[NTT];
#pragma unroll
    for (int tm = 0; tm < MT; ++tm) af[tm] = load_fragA(&As[p][wm + tm * 16 + rl][0], half);
#pragma unroll
    for (int tn = 0; tn < NTT; ++tn) bfr[tn] = load_fragB(&Bs[p][wn + tn * 16 + rl][0], half);
#pragma unroll
    for (int tm = 0; tm < MT; ++tm)
#pragma unroll
      for (int tn = 0; tn < NTT; ++tn)
        acc[tm][tn] = __builtin_amdgcn_wmma_f32_16x16x32_bf16(
            false, af[tm], false, bfr[tn], (short)0, acc[tm][tn], false, false);
    __syncthreads();
  }
  // park edge_x (+bias) in LDS for the j-reduction
  const int rr = half * 8;
#pragma unroll
  for (int tm = 0; tm < MT; ++tm)
#pragma unroll
    for (int tn = 0; tn < NTT; ++tn) {
      const int cc = wn + tn * 16 + rl;
      const float bv = ebias[col0 + cc];
#pragma unroll
      for (int r = 0; r < 8; ++r)
        eds[wm + tm * 16 + rr + r][cc] = acc[tm][tn][r] + bv;
    }
  __syncthreads();
  const int bi = blockIdx.y, b = bi >> 6;
  if (e_out) {
    for (int idx = tid; idx < 64 * BN; idx += 256) {
      const int j = idx >> 7, cc = idx & 127;
      e_out[((size_t)bi * 64 + j) * 1024 + col0 + cc] = (__bf16)fmaxf(eds[j][cc], 0.0f);
    }
  }
  if (tid < BN) {
    const int gc = col0 + tid;
    float agg = 0.0f;
#pragma unroll 4
    for (int j = 0; j < 64; ++j)
      agg += Adj[bi * 64 + j] * eds[j][tid] * nodex[(size_t)(b * 64 + j) * 1024 + gc];
    emb_out[(size_t)bi * 1024 + gc] = (__bf16)(fmaxf(agg, 0.0f) + sx[(size_t)bi * 1024 + gc]);
  }
}

// ---------------------------------------------------------------------------
// Layer-0 edge path (Ein=1): edge_x[b,i,j,c] = fe[b,i,j]*ew[c]+eb[c] (rank-1)
// ---------------------------------------------------------------------------
__global__ void __launch_bounds__(256)
edge0_kernel(const float* __restrict__ Adj, const float* __restrict__ fe,
             const float* __restrict__ ew, const float* __restrict__ eb,
             const float* __restrict__ nodex, const float* __restrict__ sx,
             __bf16* __restrict__ e_out, __bf16* __restrict__ emb_out)
{
  const int bi = blockIdx.y, b = bi >> 6;
  const int c = blockIdx.x * 256 + threadIdx.x;
  __shared__ float ar[64], fer[64];
  if (threadIdx.x < 64) {
    ar[threadIdx.x]  = Adj[bi * 64 + threadIdx.x];
    fer[threadIdx.x] = fe[bi * 64 + threadIdx.x];
  }
  __syncthreads();
  const float w = ew[c], bb = eb[c];
  float agg = 0.0f;
#pragma unroll 4
  for (int j = 0; j < 64; ++j) {
    const float ex = fer[j] * w + bb;
    e_out[((size_t)bi * 64 + j) * 1024 + c] = (__bf16)fmaxf(ex, 0.0f);
    agg += ar[j] * ex * nodex[(size_t)(b * 64 + j) * 1024 + c];
  }
  emb_out[(size_t)bi * 1024 + c] = (__bf16)(fmaxf(agg, 0.0f) + sx[(size_t)bi * 1024 + c]);
}

// ---------------------------------------------------------------------------
// Cross-graph mixing: u1 = [emb1 | s@emb2], u2 = [emb2 | s^T@emb1] (bf16, K=64)
// ---------------------------------------------------------------------------
__global__ void __launch_bounds__(256)
cross_mix_kernel(const float* __restrict__ s, const __bf16* __restrict__ emb1,
                 const __bf16* __restrict__ emb2, __bf16* __restrict__ u1,
                 __bf16* __restrict__ u2)
{
  const int bi = blockIdx.y, b = bi >> 6, i = bi & 63;
  const int c = blockIdx.x * 256 + threadIdx.x;
  __shared__ float srow[64], scol[64];
  if (threadIdx.x < 64) {
    srow[threadIdx.x] = s[b * 4096 + i * 64 + threadIdx.x];
    scol[threadIdx.x] = s[b * 4096 + threadIdx.x * 64 + i];
  }
  __syncthreads();
  float a1 = 0.0f, a2 = 0.0f;
#pragma unroll 4
  for (int j = 0; j < 64; ++j) {
    a1 += srow[j] * (float)emb2[(size_t)(b * 64 + j) * 1024 + c];
    a2 += scol[j] * (float)emb1[(size_t)(b * 64 + j) * 1024 + c];
  }
  u1[(size_t)bi * 2048 + c]        = emb1[(size_t)bi * 1024 + c];
  u1[(size_t)bi * 2048 + 1024 + c] = (__bf16)a1;
  u2[(size_t)bi * 2048 + c]        = emb2[(size_t)bi * 1024 + c];
  u2[(size_t)bi * 2048 + 1024 + c] = (__bf16)a2;
}

// ---------------------------------------------------------------------------
// Masked Sinkhorn (64x64 per batch, alternating row/col log-normalize).
// ---------------------------------------------------------------------------
__global__ void __launch_bounds__(64)
sinkhorn_kernel(const float* __restrict__ spre, float* __restrict__ out,
                const int* __restrict__ n1, const int* __restrict__ n2,
                const int* __restrict__ max_iter_p)
{
  const int b = blockIdx.x, t = threadIdx.x;
  __shared__ float ls[64][65];
  const int r1 = n1[b], r2 = n2[b];
  const bool tb = r1 > r2;
  const int nr = tb ? r2 : r1, nc = tb ? r1 : r2;
  const int max_iter = *max_iter_p;
  const float* S = spre + b * 4096;
  for (int j = 0; j < 64; ++j) {
    const float v = tb ? S[j * 64 + t] : S[t * 64 + j];
    ls[t][j] = (t < nr && j < nc) ? v * SK_INV_TAU : NEGV;
  }
  __syncthreads();
  for (int it = 0; it < max_iter; ++it) {
    if ((it & 1) == 0) {            // row-normalize: thread t owns row t
      float m = -3.4e38f;
      for (int j = 0; j < 64; ++j) m = fmaxf(m, ls[t][j]);
      float ssum = 0.0f;
      for (int j = 0; j < 64; ++j) ssum += __expf(ls[t][j] - m);
      const float lse = m + __logf(ssum);
      for (int j = 0; j < 64; ++j)
        ls[t][j] = (t < nr && j < nc) ? (ls[t][j] - lse) : NEGV;
    } else {                        // col-normalize: thread t owns col t
      float m = -3.4e38f;
      for (int i = 0; i < 64; ++i) m = fmaxf(m, ls[i][t]);
      float ssum = 0.0f;
      for (int i = 0; i < 64; ++i) ssum += __expf(ls[i][t] - m);
      const float lse = m + __logf(ssum);
      for (int i = 0; i < 64; ++i)
        ls[i][t] = (i < nr && t < nc) ? (ls[i][t] - lse) : NEGV;
    }
    __syncthreads();
  }
  float* O = out + b * 4096;
  for (int j = 0; j < 64; ++j) {
    const float v = (t < nr && j < nc) ? __expf(ls[t][j]) : 0.0f;
    if (tb) O[j * 64 + t] = v; else O[t * 64 + j] = v;
  }
}

// ---------------------------------------------------------------------------
// Utility: f32 [K][N] -> bf16 [N][K] transpose (weights), and flat f32->bf16.
// ---------------------------------------------------------------------------
__global__ void transpose_to_bf16(const float* __restrict__ W, __bf16* __restrict__ WT,
                                  int K, int N)
{
  __shared__ float tile[32][33];
  const int k0 = blockIdx.y * 32, n0 = blockIdx.x * 32;
  for (int r = threadIdx.y; r < 32; r += 8)
    tile[r][threadIdx.x] = W[(size_t)(k0 + r) * N + n0 + threadIdx.x];
  __syncthreads();
  for (int r = threadIdx.y; r < 32; r += 8)
    WT[(size_t)(n0 + r) * K + k0 + threadIdx.x] = (__bf16)tile[threadIdx.x][r];
}

__global__ void f32_to_bf16(const float* __restrict__ in, __bf16* __restrict__ out, size_t n)
{
  const size_t i = (size_t)blockIdx.x * blockDim.x + threadIdx.x;
  if (i < n) out[i] = (__bf16)in[i];
}

// ---------------------------------------------------------------------------
// Host orchestration
// ---------------------------------------------------------------------------
extern "C" void kernel_launch(void* const* d_in, const int* in_sizes, int n_in,
                              void* d_out, int out_size, void* d_ws, size_t ws_size,
                              hipStream_t stream)
{
  (void)in_sizes; (void)n_in; (void)out_size; (void)ws_size;
  const float* feat_node1 = (const float*)d_in[0];
  const float* feat_node2 = (const float*)d_in[1];
  const float* A1 = (const float*)d_in[2];
  const float* A2 = (const float*)d_in[3];
  const float* feat_edge1 = (const float*)d_in[4];
  const float* feat_edge2 = (const float*)d_in[5];
  const float *nwp[3], *nbp[3], *swp[3], *sbp[3], *ewp[3], *ebp[3];
  for (int l = 0; l < 3; ++l) {
    nwp[l] = (const float*)d_in[6 + l * 6 + 0];
    nbp[l] = (const float*)d_in[6 + l * 6 + 1];
    swp[l] = (const float*)d_in[6 + l * 6 + 2];
    sbp[l] = (const float*)d_in[6 + l * 6 + 3];
    ewp[l] = (const float*)d_in[6 + l * 6 + 4];
    ebp[l] = (const float*)d_in[6 + l * 6 + 5];
  }
  const float* aff1    = (const float*)d_in[24];
  const float* aff2    = (const float*)d_in[25];
  const float* cross_w = (const float*)d_in[26];
  const float* cross_b = (const float*)d_in[27];
  const int* n1    = (const int*)d_in[28];
  const int* n2    = (const int*)d_in[29];
  const int* skmax = (const int*)d_in[30];

  // -- workspace carve ------------------------------------------------------
  char* wsbase = (char*)d_ws;
  size_t off = 0;
  auto carve = [&](size_t bytes) -> void* {
    void* r = wsbase + off;
    off += (bytes + 255) & ~(size_t)255;
    return r;
  };
  const size_t H = 1024, NB = 512, ME = 32768;   // NB=B*N rows, ME=B*N*N edge rows
  __bf16 *nwT[3], *swT[3];
  for (int l = 0; l < 3; ++l) { nwT[l] = (__bf16*)carve(H * H * 2); swT[l] = (__bf16*)carve(H * H * 2); }
  __bf16* ewT[2];
  for (int l = 0; l < 2; ++l) ewT[l] = (__bf16*)carve(H * H * 2);
  __bf16* aff1T  = (__bf16*)carve(H * H * 2);
  __bf16* aff2T  = (__bf16*)carve(H * H * 2);
  __bf16* crossT = (__bf16*)carve(H * 2 * H * 2);
  __bf16* emb1A  = (__bf16*)carve(NB * H * 2);
  __bf16* emb1B  = (__bf16*)carve(NB * H * 2);
  __bf16* emb2A  = (__bf16*)carve(NB * H * 2);
  __bf16* emb2B  = (__bf16*)carve(NB * H * 2);
  __bf16* u1     = (__bf16*)carve(NB * 2 * H * 2);
  __bf16* u2     = (__bf16*)carve(NB * 2 * H * 2);
  __bf16* t_bf   = (__bf16*)carve(NB * H * 2);
  float*  nodex  = (float*)carve(NB * H * 4);
  float*  sxbuf  = (float*)carve(NB * H * 4);
  float*  spre   = (float*)carve(8 * 4096 * 4);
  float*  s_ws   = (float*)carve(8 * 4096 * 4);
  __bf16* e1b0 = (__bf16*)carve(ME * H * 2);
  __bf16* e1b1 = (__bf16*)carve(ME * H * 2);
  __bf16* e2b0 = (__bf16*)carve(ME * H * 2);
  __bf16* e2b1 = (__bf16*)carve(ME * H * 2);

  // -- one-time precision conversion / weight transposes --------------------
  const dim3 tblk(32, 8);
  for (int l = 0; l < 3; ++l) {
    transpose_to_bf16<<<dim3(32, 32), tblk, 0, stream>>>(nwp[l], nwT[l], 1024, 1024);
    transpose_to_bf16<<<dim3(32, 32), tblk, 0, stream>>>(swp[l], swT[l], 1024, 1024);
  }
  transpose_to_bf16<<<dim3(32, 32), tblk, 0, stream>>>(ewp[1], ewT[0], 1024, 1024);
  transpose_to_bf16<<<dim3(32, 32), tblk, 0, stream>>>(ewp[2], ewT[1], 1024, 1024);
  transpose_to_bf16<<<dim3(32, 32), tblk, 0, stream>>>(aff1, aff1T, 1024, 1024);
  transpose_to_bf16<<<dim3(32, 32), tblk, 0, stream>>>(aff2, aff2T, 1024, 1024);
  transpose_to_bf16<<<dim3(32, 64), tblk, 0, stream>>>(cross_w, crossT, 2048, 1024);
  f32_to_bf16<<<2048, 256, 0, stream>>>(feat_node1, emb1A, NB * H);
  f32_to_bf16<<<2048, 256, 0, stream>>>(feat_node2, emb2A, NB * H);

  auto gemm = [&](const __bf16* Ax, const __bf16* Btx, const float* bias,
                  float* Cf, __bf16* Cb, int M, int N, int K, int relu) {
    gemm_bf16_kernel<64, 128, 32, 32><<<dim3(N / 128, M / 64, 1), 256, 0, stream>>>(
        Ax, 0, Btx, 0, bias, Cf, Cb, 0, M, N, K, relu);
  };
  auto affinity = [&](const __bf16* T, const __bf16* E2, float* Sout) {
    gemm_bf16_kernel<64, 64, 32, 32><<<dim3(1, 1, 8), 128, 0, stream>>>(
        T, 65536, E2, 65536, nullptr, Sout, nullptr, 4096, 64, 64, 1024, 0);
  };

  __bf16 *em1_cur = emb1A, *em1_nxt = emb1B, *em2_cur = emb2A, *em2_nxt = emb2B;

  // ---- layer 0 (edge input Ein=1, rank-1 path) ----------------------------
  gemm(em1_cur, nwT[0], nbp[0], nodex, nullptr, 512, 1024, 1024, 0);
  gemm(em1_cur, swT[0], sbp[0], sxbuf, nullptr, 512, 1024, 1024, 1);
  edge0_kernel<<<dim3(4, 512), 256, 0, stream>>>(A1, feat_edge1, ewp[0], ebp[0],
                                                 nodex, sxbuf, e1b0, em1_nxt);
  gemm(em2_cur, nwT[0], nbp[0], nodex, nullptr, 512, 1024, 1024, 0);
  gemm(em2_cur, swT[0], sbp[0], sxbuf, nullptr, 512, 1024, 1024, 1);
  edge0_kernel<<<dim3(4, 512), 256, 0, stream>>>(A2, feat_edge2, ewp[0], ebp[0],
                                                 nodex, sxbuf, e2b0, em2_nxt);
  std::swap(em1_cur, em1_nxt); std::swap(em2_cur, em2_nxt);

  // ---- layer 1 ------------------------------------------------------------
  gemm(em1_cur, nwT[1], nbp[1], nodex, nullptr, 512, 1024, 1024, 0);
  gemm(em1_cur, swT[1], sbp[1], sxbuf, nullptr, 512, 1024, 1024, 1);
  edge_gemm_kernel<<<dim3(8, 512), 256, 0, stream>>>(e1b0, ewT[0], ebp[1], A1,
                                                     nodex, sxbuf, e1b1, em1_nxt, 1024);
  gemm(em2_cur, nwT[1], nbp[1], nodex, nullptr, 512, 1024, 1024, 0);
  gemm(em2_cur, swT[1], sbp[1], sxbuf, nullptr, 512, 1024, 1024, 1);
  edge_gemm_kernel<<<dim3(8, 512), 256, 0, stream>>>(e2b0, ewT[0], ebp[1], A2,
                                                     nodex, sxbuf, e2b1, em2_nxt, 1024);
  std::swap(em1_cur, em1_nxt); std::swap(em2_cur, em2_nxt);

  // ---- cross-graph Sinkhorn update (after layer index 1) ------------------
  gemm(em1_cur, aff1T, nullptr, nullptr, t_bf, 512, 1024, 1024, 0);
  affinity(t_bf, em2_cur, spre);
  sinkhorn_kernel<<<8, 64, 0, stream>>>(spre, s_ws, n1, n2, skmax);
  cross_mix_kernel<<<dim3(4, 512), 256, 0, stream>>>(s_ws, em1_cur, em2_cur, u1, u2);
  gemm(u1, crossT, cross_b, nullptr, em1_nxt, 512, 1024, 2048, 0);
  gemm(u2, crossT, cross_b, nullptr, em2_nxt, 512, 1024, 2048, 0);
  std::swap(em1_cur, em1_nxt); std::swap(em2_cur, em2_nxt);

  // ---- layer 2 (edge output dead -> skip e_out stores) --------------------
  gemm(em1_cur, nwT[2], nbp[2], nodex, nullptr, 512, 1024, 1024, 0);
  gemm(em1_cur, swT[2], sbp[2], sxbuf, nullptr, 512, 1024, 1024, 1);
  edge_gemm_kernel<<<dim3(8, 512), 256, 0, stream>>>(e1b1, ewT[1], ebp[2], A1,
                                                     nodex, sxbuf, nullptr, em1_nxt, 1024);
  gemm(em2_cur, nwT[2], nbp[2], nodex, nullptr, 512, 1024, 1024, 0);
  gemm(em2_cur, swT[2], sbp[2], sxbuf, nullptr, 512, 1024, 1024, 1);
  edge_gemm_kernel<<<dim3(8, 512), 256, 0, stream>>>(e2b1, ewT[1], ebp[2], A2,
                                                     nodex, sxbuf, nullptr, em2_nxt, 1024);
  std::swap(em1_cur, em1_nxt); std::swap(em2_cur, em2_nxt);

  // ---- final affinity + Sinkhorn -> d_out ---------------------------------
  gemm(em1_cur, aff2T, nullptr, nullptr, t_bf, 512, 1024, 1024, 0);
  affinity(t_bf, em2_cur, spre);
  sinkhorn_kernel<<<8, 64, 0, stream>>>(spre, (float*)d_out, n1, n2, skmax);
}
